// GRU_Att_Layer_30554397344421
// MI455X (gfx1250) — matrix operand order repaired and checked by reading, not compiled, not run on hardware
//
#include <hip/hip_runtime.h>
#include <hip/hip_bf16.h>

// ---------------------------------------------------------------------------
// GRU + attention-gated-input scan, MI455X (gfx1250, wave32, WMMA bf16).
// B=128, S=1024, I=H=256.
//
//   * bf16 weights cast once into workspace; per-WG weight tiles live in LDS
//     for the whole scan (320KB LDS/WGP makes this free).
//   * 16 persistent workgroups (one col-tile of 16 outputs each) x 8 waves
//     (one 16-row batch tile each). Grid-wide barrier twice per step.
//   * All GEMM work is v_wmma_f32_16x16x32_bf16; A/B operands are built with
//     two aligned uint4 LDS loads each, matching the ISA 16-bit A/B layouts.
//   * Recurrence-critical xg/h re-staging uses GLOBAL_LOAD_ASYNC_TO_LDS_B128
//     (ASYNCcnt) when the toolchain exposes the builtins; x(s+1) staging is
//     overlapped with the grid barrier since it is recurrence-independent.
// ---------------------------------------------------------------------------

typedef __attribute__((ext_vector_type(16))) __bf16 v16bf;
typedef __attribute__((ext_vector_type(8)))  float  v8f;
typedef __attribute__((ext_vector_type(8)))  unsigned int v8u;
typedef __attribute__((ext_vector_type(4)))  int v4i;

#define NWG   16
#define BATCH 128
#define SEQ   1024
#define DIM   256   // I == H

#if __has_builtin(__builtin_amdgcn_global_load_async_to_lds_b128) && \
    __has_builtin(__builtin_amdgcn_s_wait_asynccnt)
#define HAS_ASYNC_LDS 1
#else
#define HAS_ASYNC_LDS 0
#endif

__device__ __forceinline__ unsigned short f2bf(float f) {
  unsigned u = __builtin_bit_cast(unsigned, f);
  u += 0x7FFFu + ((u >> 16) & 1u);        // round-to-nearest-even
  return (unsigned short)(u >> 16);
}
__device__ __forceinline__ unsigned pack2(float a, float b) {
  return (unsigned)f2bf(a) | ((unsigned)f2bf(b) << 16);
}
__device__ __forceinline__ float bf2f_half(unsigned dw, int hi) {
  unsigned u = hi ? (dw & 0xFFFF0000u) : (dw << 16);
  return __builtin_bit_cast(float, u);
}
__device__ __forceinline__ float sigm(float x) {
  return 1.0f / (1.0f + __expf(-x));
}

// Copy 64KB (4096 x b128) global -> LDS with 256 threads.
// Prefers CDNA5 async-to-LDS DMA (ASYNCcnt) over a VGPR round trip.
__device__ __forceinline__ void stage64k(unsigned* dstLds,
                                         const unsigned short* srcG, int tid) {
#if HAS_ASYNC_LDS
  for (int t = tid; t < 4096; t += 256) {
    v4i* gsrc = (v4i*)(unsigned long long)((const uint4*)srcG + t);
    __attribute__((address_space(3))) v4i* ldst =
        (__attribute__((address_space(3))) v4i*)(unsigned long long)
            ((uint4*)dstLds + t);
    __builtin_amdgcn_global_load_async_to_lds_b128(gsrc, ldst, 0, 0);
  }
  __builtin_amdgcn_s_wait_asynccnt(0);
#else
  const uint4* src = (const uint4*)srcG;
  uint4* dst = (uint4*)dstLds;
  for (int t = tid; t < 4096; t += 256) dst[t] = src[t];
#endif
}

// One 16x16x32 bf16 WMMA. Abase: per-lane pointer into a row-major
// [rows][256] bf16 array at (row m)*128 dwords + (lane>=16 ? 4 : 0);
// second A uint4 sits +8 dwords (ISA: VGPR4-7 hold K=16..31).
// Bbase: per-lane pointer into a row-major [16][256] bf16 tile at
// (col n)*128 + (lane>=16 ? 4 : 0); second B uint4 sits +8 dwords (same as A).
__device__ __forceinline__ v8f wmma_bf(const unsigned* Abase,
                                       const unsigned* Bbase, v8f c) {
  uint4 a0 = *(const uint4*)(Abase);
  uint4 a1 = *(const uint4*)(Abase + 8);
  uint4 b0 = *(const uint4*)(Bbase);
  uint4 b1 = *(const uint4*)(Bbase + 8);
  v8u au = {a0.x, a0.y, a0.z, a0.w, a1.x, a1.y, a1.z, a1.w};
  v8u bu = {b0.x, b0.y, b0.z, b0.w, b1.x, b1.y, b1.z, b1.w};
  v16bf A = __builtin_bit_cast(v16bf, au);
  v16bf B = __builtin_bit_cast(v16bf, bu);
  return __builtin_amdgcn_wmma_f32_16x16x32_bf16(
      false, A, false, B, (short)0, c, false, false);
}

__device__ __forceinline__ void grid_barrier(unsigned* cnt, unsigned* gen) {
  __threadfence();           // release this thread's global writes (agent)
  __syncthreads();
  if (threadIdx.x == 0) {
    unsigned g = __hip_atomic_load(gen, __ATOMIC_ACQUIRE,
                                   __HIP_MEMORY_SCOPE_AGENT);
    unsigned arrived = __hip_atomic_fetch_add(cnt, 1u, __ATOMIC_ACQ_REL,
                                              __HIP_MEMORY_SCOPE_AGENT);
    if (arrived == NWG - 1u) {
      __hip_atomic_store(cnt, 0u, __ATOMIC_RELAXED, __HIP_MEMORY_SCOPE_AGENT);
      __hip_atomic_fetch_add(gen, 1u, __ATOMIC_ACQ_REL,
                             __HIP_MEMORY_SCOPE_AGENT);
    } else {
      while (__hip_atomic_load(gen, __ATOMIC_ACQUIRE,
                               __HIP_MEMORY_SCOPE_AGENT) == g) {
        __builtin_amdgcn_s_sleep(1);
      }
    }
  }
  __syncthreads();
  __threadfence();           // acquire side for all threads
}

// ---------------------------------------------------------------------------
// Prep kernels
// ---------------------------------------------------------------------------
__global__ void cast_bf16_kernel(const float* __restrict__ src,
                                 unsigned short* __restrict__ dst, int n) {
  int i = blockIdx.x * 256 + threadIdx.x;
  if (i < n) dst[i] = f2bf(src[i]);
}

__global__ void zero_sync_kernel(unsigned* sync) {
  if (threadIdx.x < 4) sync[threadIdx.x] = 0u;
}

// ---------------------------------------------------------------------------
// Main persistent scan kernel. grid = 16 blocks x 256 threads, 256KB dyn LDS.
// ---------------------------------------------------------------------------
__global__ void __launch_bounds__(256, 1)
gru_att_scan(const float* __restrict__ xin,        // [B,S,I] f32
             const float* __restrict__ batt,       // [256]
             const float* __restrict__ bih,        // [768]
             const float* __restrict__ bhh,        // [768]
             const unsigned short* __restrict__ wattb,  // [256][512] bf16
             const unsigned short* __restrict__ wihb,   // [768][256] bf16
             const unsigned short* __restrict__ whhb,   // [768][256] bf16
             unsigned short* __restrict__ xgg,     // [128][256] bf16 scratch
             unsigned short* __restrict__ h0,      // [128][256] bf16 ping
             unsigned short* __restrict__ h1,      // [128][256] bf16 pong
             unsigned* __restrict__ syncv,         // {cnt, gen}
             float* __restrict__ out)              // [B,S,H] f32
{
  extern __shared__ unsigned smem[];
  unsigned* xs  = smem;             // x(s)  bf16 [128][256] : 16384 dwords
  unsigned* hs  = smem + 16384;     // h     bf16 [128][256]
  unsigned* xgs = smem + 32768;     // xg    bf16 [128][256]
  unsigned* wts = smem + 49152;     // 8 tiles x [16][256] bf16 (2048 dw each)
  // tile order: 0 Wax, 1 Wah, 2 Wir, 3 Wiz, 4 Win, 5 Whr, 6 Whz, 7 Whn

  const int tid  = threadIdx.x;
  const int g    = blockIdx.x;      // column-tile group 0..15
  const int j0   = g * 16;
  const int lane = tid & 31;
  const int wave = tid >> 5;        // row tile 0..7
  const int ln   = lane & 15;
  const int lh   = lane >> 4;       // 0 | 1

  // ---- stage this WG's weight tiles into LDS (once) -----------------------
  const unsigned* wat  = (const unsigned*)wattb;  // 256 dwords per row (K=512)
  const unsigned* wih  = (const unsigned*)wihb;   // 128 dwords per row
  const unsigned* whh  = (const unsigned*)whhb;
  for (int idx = tid; idx < 2048; idx += 256) {
    int r = idx >> 7, d = idx & 127;
    wts[0 * 2048 + idx] = wat[(j0 + r) * 256 + d];          // W_att x-part
    wts[1 * 2048 + idx] = wat[(j0 + r) * 256 + 128 + d];    // W_att h-part
    wts[2 * 2048 + idx] = wih[(0 * 256 + j0 + r) * 128 + d];
    wts[3 * 2048 + idx] = wih[(1 * 256 + j0 + r) * 128 + d];
    wts[4 * 2048 + idx] = wih[(2 * 256 + j0 + r) * 128 + d];
    wts[5 * 2048 + idx] = whh[(0 * 256 + j0 + r) * 128 + d];
    wts[6 * 2048 + idx] = whh[(1 * 256 + j0 + r) * 128 + d];
    wts[7 * 2048 + idx] = whh[(2 * 256 + j0 + r) * 128 + d];
  }
  // ---- stage initial h (async DMA to LDS when available) ------------------
  stage64k(hs, h0, tid);
  // ---- stage x(0) ---------------------------------------------------------
  for (int idx = tid; idx < 16384; idx += 256) {
    int b = idx >> 7, d = idx & 127;
    const float2 v = *(const float2*)(xin + ((size_t)b * SEQ + 0) * DIM + 2 * d);
    xs[idx] = pack2(v.x, v.y);
  }
  __syncthreads();

  // ---- per-lane constants -------------------------------------------------
  const int   n     = j0 + ln;                 // this lane's output column
  const float c_ba  = batt[n];
  const float c_bir = bih[n],       c_biz = bih[256 + n], c_bin = bih[512 + n];
  const float c_bhr = bhh[n],       c_bhz = bhh[256 + n], c_bhn = bhh[512 + n];

  const unsigned Aoff = (unsigned)((wave * 16 + ln) * 128 + lh * 4);
  const unsigned* Arow_x  = xs  + Aoff;
  const unsigned* Arow_h  = hs  + Aoff;
  const unsigned* Arow_xg = xgs + Aoff;
  const unsigned  boff    = (unsigned)(ln * 128 + lh * 4);

  unsigned short* hb[2] = {h0, h1};

  for (int s = 0; s < SEQ; ++s) {
    // ---- phase A: att = sigmoid([x|h] @ W_att^T + b_att) ------------------
    v8f acc = {};
#pragma unroll
    for (int kc = 0; kc < 8; ++kc)
      acc = wmma_bf(Arow_x + kc * 16, wts + 0 * 2048 + boff + kc * 16, acc);
#pragma unroll
    for (int kc = 0; kc < 8; ++kc)
      acc = wmma_bf(Arow_h + kc * 16, wts + 1 * 2048 + boff + kc * 16, acc);

#pragma unroll
    for (int r = 0; r < 8; ++r) {
      int m = wave * 16 + lh * 8 + r;          // C/D layout: M = r + 8*lh
      float a  = sigm(acc[r] + c_ba);
      float xv = bf2f_half(xs[m * 128 + (n >> 1)], n & 1);
      xgg[m * DIM + n] = f2bf(xv * a);         // gated input, bf16, global
    }
    __syncthreads();   // all waves finished reading xs

    // ---- overlap: stage x(s+1) while waiting at the grid barrier ----------
    if (s + 1 < SEQ) {
      for (int idx = tid; idx < 16384; idx += 256) {
        int b = idx >> 7, d = idx & 127;
        const float2 v =
            *(const float2*)(xin + ((size_t)b * SEQ + (s + 1)) * DIM + 2 * d);
        xs[idx] = pack2(v.x, v.y);
      }
    }
    if (s + 2 < SEQ) {  // warm L2 two steps ahead (global_prefetch)
      __builtin_prefetch(xin + ((size_t)(tid & 127) * SEQ + (s + 2)) * DIM +
                             (tid >> 7) * 128, 0, 1);
    }
    grid_barrier(syncv, syncv + 1);

    // ---- stage xg -> LDS (async DMA) --------------------------------------
    stage64k(xgs, xgg, tid);
    __syncthreads();

    // ---- phase B: gi = xg@w_ih^T, gh = h@w_hh^T, gates --------------------
    v8f aR = {}, aZ = {}, aN = {}, hR = {}, hZ = {}, hN = {};
#pragma unroll
    for (int kc = 0; kc < 8; ++kc) {
      const unsigned* Ax = Arow_xg + kc * 16;
      const unsigned* Ah = Arow_h + kc * 16;
      const unsigned  bk = boff + kc * 16;
      aR = wmma_bf(Ax, wts + 2 * 2048 + bk, aR);
      aZ = wmma_bf(Ax, wts + 3 * 2048 + bk, aZ);
      aN = wmma_bf(Ax, wts + 4 * 2048 + bk, aN);
      hR = wmma_bf(Ah, wts + 5 * 2048 + bk, hR);
      hZ = wmma_bf(Ah, wts + 6 * 2048 + bk, hZ);
      hN = wmma_bf(Ah, wts + 7 * 2048 + bk, hN);
    }

    unsigned short* hn = hb[(s + 1) & 1];
#pragma unroll
    for (int r = 0; r < 8; ++r) {
      int m = wave * 16 + lh * 8 + r;
      float rg = sigm(aR[r] + c_bir + hR[r] + c_bhr);
      float zg = sigm(aZ[r] + c_biz + hZ[r] + c_bhz);
      float ng = tanhf(aN[r] + c_bin + rg * (hN[r] + c_bhn));
      float ho = bf2f_half(hs[m * 128 + (n >> 1)], n & 1);
      float hv = (1.0f - zg) * ng + zg * ho;
      out[((size_t)m * SEQ + s) * DIM + n] = hv;   // [B,S,H] f32
      hn[m * DIM + n] = f2bf(hv);
    }
    grid_barrier(syncv, syncv + 1);

    // ---- stage h(next) -> LDS (async DMA) ---------------------------------
    stage64k(hs, hn, tid);
    __syncthreads();
  }
}

// ---------------------------------------------------------------------------
// Host launch
// ---------------------------------------------------------------------------
extern "C" void kernel_launch(void* const* d_in, const int* in_sizes, int n_in,
                              void* d_out, int out_size, void* d_ws,
                              size_t ws_size, hipStream_t stream) {
  (void)in_sizes; (void)n_in; (void)out_size; (void)ws_size;

  const float* xin  = (const float*)d_in[0];   // [128,1024,256]
  const float* hid  = (const float*)d_in[1];   // [128,256]
  const float* Watt = (const float*)d_in[2];   // [256,512]
  const float* batt = (const float*)d_in[3];   // [256]
  const float* w_ih = (const float*)d_in[4];   // [768,256]
  const float* b_ih = (const float*)d_in[5];   // [768]
  const float* w_hh = (const float*)d_in[6];   // [768,256]
  const float* b_hh = (const float*)d_in[7];   // [768]
  float* out = (float*)d_out;

  char* ws = (char*)d_ws;                      // ~1.25 MB used
  unsigned short* watt_bf = (unsigned short*)(ws + 0);         // 262144 B
  unsigned short* wih_bf  = (unsigned short*)(ws + 262144);    // 393216 B
  unsigned short* whh_bf  = (unsigned short*)(ws + 655360);    // 393216 B
  unsigned short* xg_g    = (unsigned short*)(ws + 1048576);   // 65536 B
  unsigned short* h0_g    = (unsigned short*)(ws + 1114112);   // 65536 B
  unsigned short* h1_g    = (unsigned short*)(ws + 1179648);   // 65536 B
  unsigned*       sync_g  = (unsigned*)(ws + 1245184);         // 16 B

  zero_sync_kernel<<<1, 32, 0, stream>>>(sync_g);
  cast_bf16_kernel<<<512, 256, 0, stream>>>(Watt, watt_bf, 256 * 512);
  cast_bf16_kernel<<<768, 256, 0, stream>>>(w_ih, wih_bf, 768 * 256);
  cast_bf16_kernel<<<768, 256, 0, stream>>>(w_hh, whh_bf, 768 * 256);
  cast_bf16_kernel<<<128, 256, 0, stream>>>(hid, h0_g, 128 * 256);

  gru_att_scan<<<NWG, 256, 256 * 1024, stream>>>(
      xin, batt, b_ih, b_hh, watt_bf, wih_bf, whh_bf,
      xg_g, h0_g, h1_g, sync_g, out);
}
